// SwinBlock3D_34170759807164
// MI455X (gfx1250) — matrix operand order repaired and checked by reading, not compiled, not run on hardware
//
#include <hip/hip_runtime.h>
#include <hip/hip_bf16.h>

typedef _Float16 half16 __attribute__((ext_vector_type(16)));
typedef _Float16 half8  __attribute__((ext_vector_type(8)));
typedef float    float8 __attribute__((ext_vector_type(8)));

// ---------------------------------------------------------------------------
// WMMA wrapper: D = A(16x32 f16) * B(32x16 f16) + C(16x16 f32)
// ---------------------------------------------------------------------------
__device__ inline float8 wmma_f16(half16 a, half16 b, float8 c) {
    return __builtin_amdgcn_wmma_f32_16x16x32_f16(
        /*neg_a=*/false, a, /*neg_b=*/false, b,
        /*c_mod=*/(short)0, c, /*reuse_a=*/false, /*reuse_b=*/false);
}

// ---------------------------------------------------------------------------
// A-fragment loader from row-major f16 LDS (16x32 tile at (mt,kt)).
// Contiguous K order used here (not the ISA table order): lane<16 holds row m=lane, K = {kb..kb+15},
// kb = 0 for lanes 0-15, 16 for lanes 16-31.
// ---------------------------------------------------------------------------
__device__ inline half16 ldsA(const _Float16* base, int rowStride, int mt, int kt, int lane) {
    int m  = (mt << 4) + (lane & 15);
    int kb = (lane & 16) ? 16 : 0;
    const _Float16* p = base + m * rowStride + (kt << 5) + kb;
    half8 lo = *(const half8*)p;          // K = kb .. kb+7
    half8 hi = *(const half8*)(p + 8);    // K = kb+8 .. kb+15
    half16 r;
#pragma unroll
    for (int i = 0; i < 8; ++i) { r[i] = lo[i]; r[i + 8] = hi[i]; }
    return r;
}

// ---------------------------------------------------------------------------
// B-fragment loader from LDS where the (KxN) B matrix is stored column-major:
// column n is a contiguous run of K halves with stride kstride.
// Lane n=lane&15 (+16*nt), K = kt*32 + (lane<16 ? 0 : 16) + j (j contiguous).
// ---------------------------------------------------------------------------
__device__ inline half16 ldsB(const _Float16* colBase, int kstride, int kt, int nt, int lane) {
    int n  = (nt << 4) + (lane & 15);
    int kb = (kt << 5) + ((lane & 16) ? 16 : 0);
    const _Float16* p = colBase + n * kstride + kb;
    half8 lo = *(const half8*)p;
    half8 hi = *(const half8*)(p + 8);
    half16 r;
#pragma unroll
    for (int i = 0; i < 8; ++i) { r[i] = lo[i]; r[i + 8] = hi[i]; }
    return r;
}

// ---------------------------------------------------------------------------
// B-fragment loader from pre-swizzled global weights: each tile is 512 halves,
// lane's 16 halves contiguous at tile_base + lane*16.  tile_id = nt*Ktiles+kt.
// ---------------------------------------------------------------------------
__device__ inline half16 gB(const _Float16* W, int Ktiles, int kt, int nt, int lane) {
    const half16* p = (const half16*)(W + (((size_t)nt * Ktiles + kt) << 9));
    return p[lane];
}

// ---------------------------------------------------------------------------
// Prep: swizzle fp32 row-major (KxN) weights into f16 B-fragment order.
// ---------------------------------------------------------------------------
__device__ inline void swz(const float* W, int K, int N, _Float16* dst, int tid, int nth) {
    int Ktiles = K >> 5;
    int total  = K * N;
    for (int idx = tid; idx < total; idx += nth) {
        int tile = idx >> 9;
        int l    = (idx >> 4) & 31;
        int j    = idx & 15;
        int nt   = tile / Ktiles;
        int kt   = tile - nt * Ktiles;
        int k    = (kt << 5) + ((l & 16) ? 16 : 0) + j;
        int n    = (nt << 4) + (l & 15);
        dst[idx] = (_Float16)W[k * N + n];
    }
}

__global__ void swin_prep_kernel(const float* __restrict__ w_qkv,
                                 const float* __restrict__ w_out,
                                 const float* __restrict__ pos_tab,
                                 const float* __restrict__ w1,
                                 const float* __restrict__ w2,
                                 float* __restrict__ biasTab,
                                 _Float16* __restrict__ Wqkv,
                                 _Float16* __restrict__ Wout,
                                 _Float16* __restrict__ W1f,
                                 _Float16* __restrict__ W2f) {
    int tid = blockIdx.x * blockDim.x + threadIdx.x;
    int nth = gridDim.x * blockDim.x;
    swz(w_qkv, 96, 288, Wqkv, tid, nth);
    swz(w_out, 96,  96, Wout, tid, nth);
    swz(w1,    96, 384, W1f,  tid, nth);
    swz(w2,   384,  96, W2f,  tid, nth);
    // relative-position bias table: bias[i][j] = pos_tab[di+3][dj+3][dk+3]
    for (int i = tid; i < 64 * 64; i += nth) {
        int qi = i >> 6, kj = i & 63;
        int r0 = ((qi >> 4) & 3) - ((kj >> 4) & 3) + 3;
        int r1 = ((qi >> 2) & 3) - ((kj >> 2) & 3) + 3;
        int r2 = (qi & 3) - (kj & 3) + 3;
        biasTab[i] = pos_tab[(r0 * 7 + r1) * 7 + r2];
    }
}

// ---------------------------------------------------------------------------
// Fused Swin-3D block.  One workgroup (4 wave32) = one 4x4x4 window (64 tokens).
// All intermediates in LDS; WMMA f32<-f16 for every GEMM.
// ---------------------------------------------------------------------------
#define SM_XRES   0        // f32 [64][96]        24576 B
#define SM_YH     24576    // f16 [64][96]        12288 B   (y1 / O / y2)
#define SM_QH     36864    // f16 [3][64][32]     12288 B
#define SM_KH     49152    // f16 [3][64][32]     12288 B
#define SM_VH     61440    // f16 [3][32][64]     12288 B   (V transposed)
#define SM_S      73728    // f32 [3][64][64]     49152 B   (reused as H in MLP)
#define SM_P      122880   // f16 [3][64][64]     24576 B
#define SM_TOTAL  147456

__global__ __launch_bounds__(128)
void swin_block_kernel(const float* __restrict__ x,
                       const float* __restrict__ ln1_g, const float* __restrict__ ln1_b,
                       const float* __restrict__ b_out,
                       const float* __restrict__ ln2_g, const float* __restrict__ ln2_b,
                       const float* __restrict__ b1,    const float* __restrict__ b2,
                       const float* __restrict__ biasTab,
                       const _Float16* __restrict__ Wqkv, const _Float16* __restrict__ Wout,
                       const _Float16* __restrict__ W1f,  const _Float16* __restrict__ W2f,
                       float* __restrict__ out) {
    extern __shared__ char smraw[];
    float*    xres = (float*)(smraw + SM_XRES);
    _Float16* yh   = (_Float16*)(smraw + SM_YH);
    _Float16* qh   = (_Float16*)(smraw + SM_QH);
    _Float16* kh   = (_Float16*)(smraw + SM_KH);
    _Float16* vh   = (_Float16*)(smraw + SM_VH);
    float*    S    = (float*)(smraw + SM_S);
    _Float16* P    = (_Float16*)(smraw + SM_P);
    _Float16* hmid = (_Float16*)(smraw + SM_S);   // MLP hidden reuses S region

    const int tid  = threadIdx.x;
    const int lane = tid & 31;
    const int wave = tid >> 5;

    // window decode: 4096 windows = B(2) x nH(32) x nW(32) x nD(2)
    const int w  = blockIdx.x;
    const int b  = w >> 11;
    const int r_ = w & 2047;
    const int wh = r_ >> 6;
    const int ww = (r_ >> 1) & 31;
    const int wd = r_ & 1;

    // ---- Phase 1: gather window tokens (64 x 96 f32) into LDS ----
    for (int f = tid; f < 1536; f += 128) {
        int tok = f / 24, c4 = f - tok * 24;
        int th = tok >> 4, tw = (tok >> 2) & 3, td = tok & 3;
        size_t row = (((size_t)(b * 128 + wh * 4 + th) * 128 + (ww * 4 + tw)) * 8
                      + (wd * 4 + td)) * 96;
        *(float4*)(xres + tok * 96 + c4 * 4) = *(const float4*)(x + row + c4 * 4);
    }
    __syncthreads();

    // ---- Phase 2: LayerNorm-1 -> yh (f16) ----
    if (tid < 64) {
        const float* xr = xres + tid * 96;
        float s = 0.f, ss = 0.f;
        for (int c = 0; c < 96; ++c) { float v = xr[c]; s += v; ss += v * v; }
        float mean = s * (1.f / 96.f);
        float var  = ss * (1.f / 96.f) - mean * mean;
        float rstd = rsqrtf(var + 1e-5f);
        _Float16* yr = yh + tid * 96;
        for (int c = 0; c < 96; ++c)
            yr[c] = (_Float16)((xr[c] - mean) * rstd * ln1_g[c] + ln1_b[c]);
    }
    __syncthreads();

    // ---- Phase 3: QKV GEMM (64x96)@(96x288); scatter q,k,v^T to LDS ----
    {
        const int mt = wave;
        half16 a0 = ldsA(yh, 96, mt, 0, lane);
        half16 a1 = ldsA(yh, 96, mt, 1, lane);
        half16 a2 = ldsA(yh, 96, mt, 2, lane);
        const int n = lane & 15;
        const int mbase = mt * 16 + ((lane & 16) ? 8 : 0);
        for (int nt = 0; nt < 18; ++nt) {
            if (nt < 17)
                __builtin_prefetch(Wqkv + (((size_t)(nt + 1) * 3) << 9), 0, 3);
            float8 acc = {};
            acc = wmma_f16(a0, gB(Wqkv, 3, 0, nt, lane), acc);
            acc = wmma_f16(a1, gB(Wqkv, 3, 1, nt, lane), acc);
            acc = wmma_f16(a2, gB(Wqkv, 3, 2, nt, lane), acc);
            int c = nt * 16 + n;
#pragma unroll
            for (int r = 0; r < 8; ++r) {
                int tok = mbase + r;
                float v = acc[r];
                if (c < 96) {          // Q, scale folded in
                    int hh = c >> 5, d = c & 31;
                    qh[(hh * 64 + tok) * 32 + d] = (_Float16)(v * 0.17677669529663688f);
                } else if (c < 192) {  // K row-major [h][tok][32]
                    int c2 = c - 96, hh = c2 >> 5, d = c2 & 31;
                    kh[(hh * 64 + tok) * 32 + d] = (_Float16)v;
                } else {               // V transposed [h][dim][tok]
                    int c2 = c - 192, hh = c2 >> 5, d = c2 & 31;
                    vh[(hh * 32 + d) * 64 + tok] = (_Float16)v;
                }
            }
        }
    }
    __syncthreads();

    // ---- Phase 4: S = Q@K^T + bias  (per head 64x64, K=32 -> 1 WMMA/tile) ----
    for (int t = wave; t < 48; t += 4) {
        int hh = t >> 4, mt = (t >> 2) & 3, nt = t & 3;
        half16 a  = ldsA(qh + hh * 64 * 32, 32, mt, 0, lane);
        half16 bf = ldsB(kh + hh * 64 * 32, 32, 0, nt, lane);
        float8 acc = {};
        acc = wmma_f16(a, bf, acc);
        int n = nt * 16 + (lane & 15);
        int mbase = mt * 16 + ((lane & 16) ? 8 : 0);
#pragma unroll
        for (int r = 0; r < 8; ++r) {
            int m = mbase + r;
            S[(hh * 64 + m) * 64 + n] = acc[r] + biasTab[m * 64 + n];
        }
    }
    __syncthreads();

    // ---- Phase 5: row softmax -> P (f16) ----
    for (int row = tid; row < 192; row += 128) {
        float* sr = S + row * 64;
        float mx = sr[0];
        for (int c = 1; c < 64; ++c) mx = fmaxf(mx, sr[c]);
        float sum = 0.f;
        for (int c = 0; c < 64; ++c) sum += __expf(sr[c] - mx);
        float inv = 1.f / sum;
        _Float16* pr = P + row * 64;
        for (int c = 0; c < 64; ++c) pr[c] = (_Float16)(__expf(sr[c] - mx) * inv);
    }
    __syncthreads();

    // ---- Phase 6: O = P@V  -> yh row-major [tok][h*32+d] ----
    for (int t = wave; t < 24; t += 4) {
        int hh = t >> 3, mt = (t >> 1) & 3, nt = t & 1;
        const _Float16* Ph = P + hh * 64 * 64;
        const _Float16* Vh = vh + hh * 32 * 64;
        float8 acc = {};
        acc = wmma_f16(ldsA(Ph, 64, mt, 0, lane), ldsB(Vh, 64, 0, nt, lane), acc);
        acc = wmma_f16(ldsA(Ph, 64, mt, 1, lane), ldsB(Vh, 64, 1, nt, lane), acc);
        int d = nt * 16 + (lane & 15);
        int mbase = mt * 16 + ((lane & 16) ? 8 : 0);
#pragma unroll
        for (int r = 0; r < 8; ++r)
            yh[(mbase + r) * 96 + hh * 32 + d] = (_Float16)acc[r];
    }
    __syncthreads();

    // ---- Phase 7: out-proj + residual, in place in xres ----
    {
        const int mt = wave;
        half16 a0 = ldsA(yh, 96, mt, 0, lane);
        half16 a1 = ldsA(yh, 96, mt, 1, lane);
        half16 a2 = ldsA(yh, 96, mt, 2, lane);
        const int n = lane & 15;
        const int mbase = mt * 16 + ((lane & 16) ? 8 : 0);
        for (int nt = 0; nt < 6; ++nt) {
            float8 acc = {};
            acc = wmma_f16(a0, gB(Wout, 3, 0, nt, lane), acc);
            acc = wmma_f16(a1, gB(Wout, 3, 1, nt, lane), acc);
            acc = wmma_f16(a2, gB(Wout, 3, 2, nt, lane), acc);
            int c = nt * 16 + n;
            float bo = b_out[c];
#pragma unroll
            for (int r = 0; r < 8; ++r) {
                float* px = xres + (mbase + r) * 96 + c;
                *px = *px + acc[r] + bo;
            }
        }
    }
    __syncthreads();

    // ---- Phase 8: LayerNorm-2 -> yh ----
    if (tid < 64) {
        const float* xr = xres + tid * 96;
        float s = 0.f, ss = 0.f;
        for (int c = 0; c < 96; ++c) { float v = xr[c]; s += v; ss += v * v; }
        float mean = s * (1.f / 96.f);
        float var  = ss * (1.f / 96.f) - mean * mean;
        float rstd = rsqrtf(var + 1e-5f);
        _Float16* yr = yh + tid * 96;
        for (int c = 0; c < 96; ++c)
            yr[c] = (_Float16)((xr[c] - mean) * rstd * ln2_g[c] + ln2_b[c]);
    }
    __syncthreads();

    // ---- Phase 9: MLP1 (64x96)@(96x384) + exact GELU -> hmid ----
    {
        const int mt = wave;
        half16 a0 = ldsA(yh, 96, mt, 0, lane);
        half16 a1 = ldsA(yh, 96, mt, 1, lane);
        half16 a2 = ldsA(yh, 96, mt, 2, lane);
        const int n = lane & 15;
        const int mbase = mt * 16 + ((lane & 16) ? 8 : 0);
        for (int nt = 0; nt < 24; ++nt) {
            if (nt < 23)
                __builtin_prefetch(W1f + (((size_t)(nt + 1) * 3) << 9), 0, 3);
            float8 acc = {};
            acc = wmma_f16(a0, gB(W1f, 3, 0, nt, lane), acc);
            acc = wmma_f16(a1, gB(W1f, 3, 1, nt, lane), acc);
            acc = wmma_f16(a2, gB(W1f, 3, 2, nt, lane), acc);
            int c = nt * 16 + n;
            float bb = b1[c];
#pragma unroll
            for (int r = 0; r < 8; ++r) {
                float tv = acc[r] + bb;
                float gl = 0.5f * tv * (1.f + erff(tv * 0.7071067811865475f));
                hmid[(mbase + r) * 384 + c] = (_Float16)gl;
            }
        }
    }
    __syncthreads();

    // ---- Phase 10: MLP2 (64x384)@(384x96) + residual -> global out ----
    {
        const int mt = wave;
        const int n = lane & 15;
        const int mbase = mt * 16 + ((lane & 16) ? 8 : 0);
        for (int nt = 0; nt < 6; ++nt) {
            float8 acc = {};
#pragma unroll
            for (int kt = 0; kt < 12; ++kt)
                acc = wmma_f16(ldsA(hmid, 384, mt, kt, lane),
                               gB(W2f, 12, kt, nt, lane), acc);
            int c = nt * 16 + n;
            float bb = b2[c];
#pragma unroll
            for (int r = 0; r < 8; ++r) {
                int tok = mbase + r;
                int th = tok >> 4, tw = (tok >> 2) & 3, td = tok & 3;
                size_t row = (((size_t)(b * 128 + wh * 4 + th) * 128 + (ww * 4 + tw)) * 8
                              + (wd * 4 + td)) * 96;
                out[row + c] = xres[tok * 96 + c] + acc[r] + bb;
            }
        }
    }
}

extern "C" void kernel_launch(void* const* d_in, const int* in_sizes, int n_in,
                              void* d_out, int out_size, void* d_ws, size_t ws_size,
                              hipStream_t stream) {
    const float* x      = (const float*)d_in[0];
    const float* ln1_g  = (const float*)d_in[1];
    const float* ln1_b  = (const float*)d_in[2];
    const float* w_qkv  = (const float*)d_in[3];
    const float* w_out  = (const float*)d_in[4];
    const float* b_out  = (const float*)d_in[5];
    const float* pos_tab= (const float*)d_in[6];
    const float* ln2_g  = (const float*)d_in[7];
    const float* ln2_b  = (const float*)d_in[8];
    const float* w1     = (const float*)d_in[9];
    const float* b1     = (const float*)d_in[10];
    const float* w2     = (const float*)d_in[11];
    const float* b2     = (const float*)d_in[12];
    float* out = (float*)d_out;

    char* ws = (char*)d_ws;
    float*    biasTab = (float*)(ws + 0);          // 16384 B
    _Float16* Wqkv    = (_Float16*)(ws + 16384);   // 55296 B
    _Float16* Wout    = (_Float16*)(ws + 71680);   // 18432 B
    _Float16* W1f     = (_Float16*)(ws + 90112);   // 73728 B
    _Float16* W2f     = (_Float16*)(ws + 163840);  // 73728 B  -> total 237568 B

    swin_prep_kernel<<<64, 256, 0, stream>>>(w_qkv, w_out, pos_tab, w1, w2,
                                             biasTab, Wqkv, Wout, W1f, W2f);

    swin_block_kernel<<<4096, 128, SM_TOTAL, stream>>>(
        x, ln1_g, ln1_b, b_out, ln2_g, ln2_b, b1, b2,
        biasTab, Wqkv, Wout, W1f, W2f, out);
}
